// LSTM_ED_17935783428647
// MI455X (gfx1250) — matrix operand order repaired and checked by reading, not compiled, not run on hardware
//
#include <hip/hip_runtime.h>
#include <hip/hip_bf16.h>

// ---------------------------------------------------------------------------
// LSTM encoder-decoder for MI455X (gfx1250, wave32, WMMA bf16 16x16x32,
// async global->LDS staging with ASYNCcnt double buffering).
// B=256, TE=336, TD=24, F=64, H=1024, TOT=64
// ---------------------------------------------------------------------------

#define B_   256
#define TE_  336
#define TD_  24
#define F_   64
#define H_   1024
#define TOT_ 64
#define G4H_ 4096

typedef __bf16 bf16;
typedef __bf16 bf16x8 __attribute__((ext_vector_type(8)));
typedef __bf16 v16bf  __attribute__((ext_vector_type(16)));
typedef float  v8f    __attribute__((ext_vector_type(8)));

__device__ __forceinline__ float sigf(float x) { return 1.0f / (1.0f + __expf(-x)); }

// --- CDNA5 async copy: 16 bytes global -> LDS, tracked by ASYNCcnt ----------
// VDST = VGPR holding wave-relative LDS byte address (low 32 bits of generic
// LDS pointer), VADDR = 64-bit global address, saddr = off.
__device__ __forceinline__ void async_b128(const bf16* g, bf16* l) {
    unsigned lds = (unsigned)(size_t)l;
    unsigned long long ga = (unsigned long long)(size_t)g;
    asm volatile("global_load_async_to_lds_b128 %0, %1, off"
                 :: "v"(lds), "v"(ga) : "memory");
}
__device__ __forceinline__ void wait_async_le6() {
    asm volatile("s_wait_asynccnt 0x6" ::: "memory");
}
__device__ __forceinline__ void wait_async_0() {
    asm volatile("s_wait_asynccnt 0x0" ::: "memory");
}

// Build a 16-elem bf16 fragment from two 16B LDS reads (ds_load_b128 pair).
__device__ __forceinline__ v16bf frag_lds(const bf16* p, int hi_off) {
    bf16x8 lo = *(const bf16x8*)(p);
    bf16x8 hi = *(const bf16x8*)(p + hi_off);
    return __builtin_shufflevector(lo, hi, 0,1,2,3,4,5,6,7,8,9,10,11,12,13,14,15);
}

// A fragment (16x32) direct from global, row-major lda. ISA layout:
// lane L<16: row L, K = k0..k0+7, k0+16..k0+23 ; lane L>=16: row L-16, K+8/+24.
__device__ __forceinline__ v16bf load_a16x32_g(const bf16* __restrict__ A, int lda,
                                               int row0, int k0, int lane) {
    const int r  = row0 + (lane & 15);
    const int kb = k0 + ((lane >> 4) << 3);
    const bf16* p = A + (size_t)r * (size_t)lda + kb;
    return frag_lds(p, 16);
}
// B fragment (32x16) from pre-transposed Bt[N x K]: same K order as the A fragment.
__device__ __forceinline__ v16bf load_b32x16_g(const bf16* __restrict__ Bt, int ldb,
                                               int col0, int k0, int lane) {
    const int c  = col0 + (lane & 15);
    const int kb = k0 + ((lane >> 4) << 3);
    const bf16* p = Bt + (size_t)c * (size_t)ldb + kb;
    return frag_lds(p, 16);
}

// ---------------------------------------------------------------------------
// Fused LSTM step with async-LDS double buffering.
// Grid (B/64, H/32), block 128 threads = 4 waves (2 row-groups x 2 col-groups).
// Block tile: 64 rows x 32 hidden cols x 4 gates.
// Wave tile : 32 rows x 16 cols x 4 gates -> 8 accumulators, 8 WMMA/chunk,
//             each B fragment reused for 2 A fragments.
// 34 K-chunks of 32: 2 chunks of [x_t, Wt] (K=F=64) + 32 chunks of [h, Ut].
// LDS rows padded to 40 bf16 (80B): 16B reads from 16 consecutive rows hit
// distinct bank groups (20*r mod 64) -> conflict-free ds_load_b128.
// ---------------------------------------------------------------------------
__global__ __launch_bounds__(128)
void lstm_step_kernel(const bf16* __restrict__ x_t, int ldx,
                      const bf16* __restrict__ h_in, int ldh_in,
                      const bf16* __restrict__ Wt,    // [4H x F]  bf16
                      const bf16* __restrict__ Ut,    // [4H x H]  bf16
                      const float* __restrict__ bias, // [4H] fp32
                      float* __restrict__ cst,        // [B x H] fp32 cell state
                      bf16*  __restrict__ h_out, int ldh_out)
{
    __shared__ __align__(16) bf16 Asm[2][64][40];    // 64 rows  x 32 K (padded)
    __shared__ __align__(16) bf16 Bsm[2][128][40];   // 4g*32col x 32 K (padded)

    const int t     = threadIdx.x;
    const int lane  = t & 31;
    const int wid   = t >> 5;
    const int wm    = wid & 1;     // row group: rows 32*wm .. +32
    const int wn    = wid >> 1;    // col group: cols 16*wn .. +16
    const int row0b = blockIdx.x * 64;
    const int col0b = blockIdx.y * 32;
    const int n     = col0b + 16 * wn + (lane & 15);

    v8f acc[4][2];
    #pragma unroll
    for (int g = 0; g < 4; ++g) {
        const float bv = bias[g * H_ + n];
        acc[g][0] = (v8f){bv, bv, bv, bv, bv, bv, bv, bv};
        acc[g][1] = acc[g][0];
    }

    const int NCH = 2 + H_ / 32;   // 34 chunks

    // Issue the 6 async copies (2 A segs + 4 B segs per thread) for chunk i
    // into LDS buffer s. Chunks 0..1 source [x_t, Wt], 2.. source [h, Ut].
    auto issue = [&](int s, int i) {
        const bf16* Ag; size_t lda; const bf16* Bg; size_t ldb;
        if (i < 2) {
            Ag = x_t + (size_t)row0b * (size_t)ldx + i * 32;  lda = (size_t)ldx;
            Bg = Wt + i * 32;                                 ldb = (size_t)F_;
        } else {
            const int k0 = (i - 2) * 32;
            Ag = h_in + (size_t)row0b * (size_t)ldh_in + k0;  lda = (size_t)ldh_in;
            Bg = Ut + k0;                                     ldb = (size_t)H_;
        }
        #pragma unroll
        for (int j = 0; j < 2; ++j) {            // A: 64 rows * 4 x 16B = 256 segs
            const int seg = t + j * 128;
            const int row = seg >> 2;
            const int so  = (seg & 3) * 8;       // bf16 elements (16B granules)
            async_b128(Ag + (size_t)row * lda + so, &Asm[s][row][so]);
        }
        #pragma unroll
        for (int j = 0; j < 4; ++j) {            // B: 128 gate-cols * 4 x 16B
            const int seg = t + j * 128;
            const int gc  = seg >> 2;            // g*32 + c
            const int so  = (seg & 3) * 8;
            const int g   = gc >> 5;
            const int c   = gc & 31;
            async_b128(Bg + (size_t)(g * H_ + col0b + c) * ldb + so, &Bsm[s][gc][so]);
        }
    };

    issue(0, 0);
    for (int i = 0; i < NCH; ++i) {
        const int cur = i & 1;
        if (i + 1 < NCH) { issue(cur ^ 1, i + 1); wait_async_le6(); }
        else             { wait_async_0(); }
        __syncthreads();   // chunk i staged by all waves

        v16bf a0 = frag_lds(&Asm[cur][32 * wm +      (lane & 15)][(lane >> 4) << 3], 16);
        v16bf a1 = frag_lds(&Asm[cur][32 * wm + 16 + (lane & 15)][(lane >> 4) << 3], 16);
        #pragma unroll
        for (int g = 0; g < 4; ++g) {
            v16bf b = frag_lds(&Bsm[cur][g * 32 + 16 * wn + (lane & 15)][(lane >> 4) << 3], 16);
            acc[g][0] = __builtin_amdgcn_wmma_f32_16x16x32_bf16(
                false, a0, false, b, (short)0, acc[g][0], false, false);
            acc[g][1] = __builtin_amdgcn_wmma_f32_16x16x32_bf16(
                false, a1, false, b, (short)0, acc[g][1], false, false);
        }
        __syncthreads();   // all waves done reading buf[cur] before reuse
    }

    // ---- gates + state update (C/D layout: m = e + 8*(lane>>4), n = lane&15) ----
    const int half = (lane >> 4) << 3;
    #pragma unroll
    for (int tm = 0; tm < 2; ++tm) {
        #pragma unroll
        for (int e = 0; e < 8; ++e) {
            const int m = row0b + 32 * wm + 16 * tm + half + e;
            const float zi = acc[0][tm][e];
            const float zf = acc[1][tm][e];
            const float zg = acc[2][tm][e];
            const float zo = acc[3][tm][e];
            const size_t ci = (size_t)m * H_ + n;
            const float cp = cst[ci];
            const float cn = sigf(zf) * cp + sigf(zi) * fmaxf(zg, 0.0f);
            const float hn = sigf(zo) * fmaxf(cn, 0.0f);
            cst[ci] = cn;
            h_out[(size_t)m * (size_t)ldh_out + n] = (bf16)hn;
        }
    }
}

// ---------------------------------------------------------------------------
// Generic WMMA GEMM with bias + optional relu: out = act(A @ Bt^T + bias).
// A: [M x K] bf16 row-major (lda), Bt: [N x K] bf16 (ldb = K).
// Grid: (M/64, N/32), block 256. Writes bf16 and/or fp32.
// ---------------------------------------------------------------------------
__global__ __launch_bounds__(256)
void gemm_bias_act_kernel(const bf16* __restrict__ A, int lda,
                          const bf16* __restrict__ Bt, int ldb,
                          const float* __restrict__ bias,
                          float* __restrict__ outF, bf16* __restrict__ outB, int ldo,
                          int K, int relu)
{
    const int lane = threadIdx.x & 31;
    const int wid  = threadIdx.x >> 5;
    const int wm   = wid & 3;
    const int wn   = wid >> 2;
    const int row0 = blockIdx.x * 64 + wm * 16;
    const int col0 = blockIdx.y * 32 + wn * 16;
    const int n    = col0 + (lane & 15);

    const float bv = bias[n];
    v8f acc = (v8f){bv, bv, bv, bv, bv, bv, bv, bv};

    for (int k0 = 0; k0 < K; k0 += 32) {
        v16bf a = load_a16x32_g(A, lda, row0, k0, lane);
        v16bf b = load_b32x16_g(Bt, ldb, col0, k0, lane);
        acc = __builtin_amdgcn_wmma_f32_16x16x32_bf16(
            false, a, false, b, (short)0, acc, false, false);
    }

    const int half = (lane >> 4) << 3;
    #pragma unroll
    for (int e = 0; e < 8; ++e) {
        const int m = row0 + half + e;
        float v = acc[e];
        if (relu) v = fmaxf(v, 0.0f);
        if (outB) outB[(size_t)m * (size_t)ldo + n] = (bf16)v;
        if (outF) outF[(size_t)m * (size_t)ldo + n] = v;
    }
}

// ---------------------------------------------------------------------------
// Prep kernels
// ---------------------------------------------------------------------------
__global__ void f32_to_bf16_kernel(const float* __restrict__ in,
                                   bf16* __restrict__ out, int n) {
    int i = blockIdx.x * blockDim.x + threadIdx.x;
    if (i < n) out[i] = (bf16)in[i];
}

// out[n*K + k] = (bf16) in[k*N + n]   (in: [K x N] fp32, out: [N x K] bf16)
__global__ void transpose_f32_to_bf16_kernel(const float* __restrict__ in,
                                             bf16* __restrict__ out, int K, int N) {
    int i = blockIdx.x * blockDim.x + threadIdx.x;
    if (i < K * N) {
        int nn = i / K;
        int kk = i - nn * K;
        out[(size_t)nn * K + kk] = (bf16)in[(size_t)kk * N + nn];
    }
}

__global__ void init_state_kernel(bf16* __restrict__ h, float* __restrict__ c, int n) {
    int i = blockIdx.x * blockDim.x + threadIdx.x;
    if (i < n) { h[i] = (bf16)0.0f; c[i] = 0.0f; }
}

// ---------------------------------------------------------------------------
// Host launcher
// ---------------------------------------------------------------------------
extern "C" void kernel_launch(void* const* d_in, const int* in_sizes, int n_in,
                              void* d_out, int out_size, void* d_ws, size_t ws_size,
                              hipStream_t stream)
{
    (void)in_sizes; (void)n_in; (void)out_size; (void)ws_size;

    const float* inp_e = (const float*)d_in[0];
    const float* inp_d = (const float*)d_in[1];
    const float* We    = (const float*)d_in[2];
    const float* Ue    = (const float*)d_in[3];
    const float* be    = (const float*)d_in[4];
    const float* Wd    = (const float*)d_in[5];
    const float* Ud    = (const float*)d_in[6];
    const float* bd    = (const float*)d_in[7];
    const float* W0    = (const float*)d_in[8];
    const float* b0    = (const float*)d_in[9];
    const float* W1    = (const float*)d_in[10];
    const float* b1    = (const float*)d_in[11];
    float* out = (float*)d_out;

    // workspace carve-up (256B aligned)
    uint8_t* ws = (uint8_t*)d_ws;
    size_t off = 0;
    auto alloc = [&](size_t bytes) -> void* {
        void* p = ws + off;
        off += (bytes + 255) & ~(size_t)255;
        return p;
    };

    bf16*  xe   = (bf16*) alloc((size_t)B_ * TE_ * F_ * 2);
    bf16*  xd   = (bf16*) alloc((size_t)B_ * TD_ * F_ * 2);
    bf16*  Wte  = (bf16*) alloc((size_t)G4H_ * F_ * 2);
    bf16*  Ute  = (bf16*) alloc((size_t)G4H_ * H_ * 2);
    bf16*  Wtd  = (bf16*) alloc((size_t)G4H_ * F_ * 2);
    bf16*  Utd  = (bf16*) alloc((size_t)G4H_ * H_ * 2);
    bf16*  W0t  = (bf16*) alloc((size_t)H_ * H_ * 2);
    bf16*  W1t  = (bf16*) alloc((size_t)TOT_ * H_ * 2);
    bf16*  hA   = (bf16*) alloc((size_t)B_ * H_ * 2);
    bf16*  hB   = (bf16*) alloc((size_t)B_ * H_ * 2);
    float* cst  = (float*)alloc((size_t)B_ * H_ * 4);
    bf16*  hseq = (bf16*) alloc((size_t)B_ * TD_ * H_ * 2);
    bf16*  o0   = (bf16*) alloc((size_t)B_ * TD_ * H_ * 2);

    auto cdiv = [](int a, int b) { return (a + b - 1) / b; };

    // ---- prep: fp32 -> bf16 (+ weight transposes so B-fragments are contiguous-K) ----
    f32_to_bf16_kernel<<<cdiv(B_ * TE_ * F_, 256), 256, 0, stream>>>(inp_e, xe, B_ * TE_ * F_);
    f32_to_bf16_kernel<<<cdiv(B_ * TD_ * F_, 256), 256, 0, stream>>>(inp_d, xd, B_ * TD_ * F_);
    transpose_f32_to_bf16_kernel<<<cdiv(F_ * G4H_, 256), 256, 0, stream>>>(We, Wte, F_, G4H_);
    transpose_f32_to_bf16_kernel<<<cdiv(H_ * G4H_, 256), 256, 0, stream>>>(Ue, Ute, H_, G4H_);
    transpose_f32_to_bf16_kernel<<<cdiv(F_ * G4H_, 256), 256, 0, stream>>>(Wd, Wtd, F_, G4H_);
    transpose_f32_to_bf16_kernel<<<cdiv(H_ * G4H_, 256), 256, 0, stream>>>(Ud, Utd, H_, G4H_);
    transpose_f32_to_bf16_kernel<<<cdiv(H_ * H_, 256), 256, 0, stream>>>(W0, W0t, H_, H_);
    transpose_f32_to_bf16_kernel<<<cdiv(H_ * TOT_, 256), 256, 0, stream>>>(W1, W1t, H_, TOT_);
    init_state_kernel<<<cdiv(B_ * H_, 256), 256, 0, stream>>>(hA, cst, B_ * H_);

    const dim3 sgrid(B_ / 64, H_ / 32);   // (4, 32) = 128 blocks
    const dim3 sblk(128);

    // ---- encoder: 336 sequential fused steps ----
    bf16* hc = hA;
    bf16* hn = hB;
    for (int t = 0; t < TE_; ++t) {
        lstm_step_kernel<<<sgrid, sblk, 0, stream>>>(
            xe + (size_t)t * F_, TE_ * F_,
            hc, H_,
            Wte, Ute, be, cst,
            hn, H_);
        bf16* tmp = hc; hc = hn; hn = tmp;
    }

    // ---- decoder: writes h_t straight into [B, TD, H] sequence buffer ----
    for (int t = 0; t < TD_; ++t) {
        const bf16* hin = (t == 0) ? hc : (hseq + (size_t)(t - 1) * H_);
        const int ldin  = (t == 0) ? H_ : (TD_ * H_);
        lstm_step_kernel<<<sgrid, sblk, 0, stream>>>(
            xd + (size_t)t * F_, TD_ * F_,
            hin, ldin,
            Wtd, Utd, bd, cst,
            hseq + (size_t)t * H_, TD_ * H_);
    }

    // ---- dense0 (relu) + dense head ----
    const int M = B_ * TD_;   // 6144
    gemm_bias_act_kernel<<<dim3(M / 64, H_ / 32), 256, 0, stream>>>(
        hseq, H_, W0t, H_, b0, nullptr, o0, H_, H_, 1);
    gemm_bias_act_kernel<<<dim3(M / 64, TOT_ / 32), 256, 0, stream>>>(
        o0, H_, W1t, H_, b1, out, nullptr, TOT_, H_, 0);
}
